// SparseTopKCoCN_76459007803897
// MI455X (gfx1250) — compile-verified, matrix operand order checked by reading, not run on hardware
//
#include <hip/hip_runtime.h>
#include <hip/hip_bf16.h>
#include <math.h>

// Problem constants (match reference)
#define BS   8
#define NN   4096
#define DF   128
#define NH   4
#define NB   32
#define KBK  128           // block size k = NN/NB
#define DH   256
#define NC   16
#define NTOT (BS*NN)       // 32768
#define NBLKS (NH*BS*NB)   // 1024

typedef __attribute__((ext_vector_type(16))) __bf16 v16bf;
typedef __attribute__((ext_vector_type(8)))  float  v8f;

__device__ __forceinline__ unsigned short f2bf_u(float f) {
  unsigned u = __builtin_bit_cast(unsigned, f);
  unsigned r = u + 0x7FFFu + ((u >> 16) & 1u);   // round-to-nearest-even
  return (unsigned short)(r >> 16);
}
__device__ __forceinline__ __bf16 f2bf(float f) {
  unsigned short h = f2bf_u(f);
  return __builtin_bit_cast(__bf16, h);
}

// ---------------- Stage 1: degree ----------------
__global__ void k_init_deg(float* deg) {
  int t = blockIdx.x * blockDim.x + threadIdx.x;
  if (t < NTOT) deg[t] = 1.0f;
}
__global__ void k_deg_scatter(const int* __restrict__ dst, float* deg, int E) {
  int t = blockIdx.x * blockDim.x + threadIdx.x;
  if (t < E) atomicAdd(&deg[dst[t]], 1.0f);
}

// -------- Stage 2: x @ W_rank, and diagonal term of scores --------
__global__ void k_rank(const float* __restrict__ x, const float* __restrict__ Wr,
                       const float* __restrict__ deg, float* xr, float* scs) {
  int t = blockIdx.x * blockDim.x + threadIdx.x;   // NTOT*NH
  int n = t >> 2, h = t & 3;
  const float* row = x + (size_t)n * DF;
  float a = 0.f;
  #pragma unroll 4
  for (int c = 0; c < DF; ++c) a += row[c] * Wr[c * NH + h];
  xr[t]  = a;
  scs[t] = a / deg[n];        // diagonal v/deg term of ahat_mv
}

__global__ void k_edge_w(const int* __restrict__ src, const int* __restrict__ dst,
                         const float* __restrict__ deg, float* wE, int E) {
  int t = blockIdx.x * blockDim.x + threadIdx.x;
  if (t < E) wE[t] = rsqrtf(deg[src[t]] * deg[dst[t]]);
}

// -------- Stage 3: x_appd = x/deg, then += w * x[src] (scatter) --------
__global__ void k_init_appd(const float* __restrict__ x, const float* __restrict__ deg,
                            float* xa) {
  int fi = blockIdx.x * blockDim.x + threadIdx.x;  // NTOT*DF/4 float4s
  int n = fi >> 5;                                 // 32 float4 per row
  float4 f = ((const float4*)x)[fi];
  float inv = 1.0f / deg[n];
  float4 o; o.x = f.x * inv; o.y = f.y * inv; o.z = f.z * inv; o.w = f.w * inv;
  ((float4*)xa)[fi] = o;
}
__global__ void k_edge_feat(const int* __restrict__ src, const int* __restrict__ dst,
                            const float* __restrict__ wE, const float* __restrict__ x,
                            float* xa, int E) {
  int t = blockIdx.x * blockDim.x + threadIdx.x;   // E*32
  int e = t >> 5, l = t & 31;
  int s = src[e], d = dst[e];
  float w = wE[e];
  float4 f = *(const float4*)(x + (size_t)s * DF + l * 4);
  float* o = xa + (size_t)d * DF + l * 4;
  atomicAdd(o + 0, w * f.x); atomicAdd(o + 1, w * f.y);
  atomicAdd(o + 2, w * f.z); atomicAdd(o + 3, w * f.w);
}
__global__ void k_edge_score(const int* __restrict__ src, const int* __restrict__ dst,
                             const float* __restrict__ wE, const float* __restrict__ xr,
                             float* scs, int E) {
  int e = blockIdx.x * blockDim.x + threadIdx.x;
  if (e >= E) return;
  int s = src[e], d = dst[e];
  float w = wE[e];
  #pragma unroll
  for (int h = 0; h < NH; ++h)
    atomicAdd(&scs[d * NH + h], w * xr[s * NH + h]);
}

// -------- Stage 4: per-(b,h) bitonic sort of 4096 scores (descending) --------
__global__ void k_sort(const float* __restrict__ scs, int* order, int* pos, float* val) {
  __shared__ float key[NN];
  __shared__ int   sid[NN];
  int bid = blockIdx.x;                 // b*NH + h
  int b = bid >> 2, h = bid & 3;
  int tid = threadIdx.x;                // 1024 threads
  #pragma unroll
  for (int q = 0; q < 4; ++q) {
    int p = tid + q * 1024;
    key[p] = scs[(size_t)(b * NN + p) * NH + h];
    sid[p] = p;
  }
  __syncthreads();
  for (int len = 2; len <= NN; len <<= 1) {
    for (int str = len >> 1; str > 0; str >>= 1) {
      #pragma unroll
      for (int q = 0; q < 2; ++q) {
        int p = tid + q * 1024;                       // 2048 pair slots
        int i = ((p & ~(str - 1)) << 1) | (p & (str - 1));
        int jj = i + str;
        bool descBlk = ((i & len) == 0);
        float ki = key[i], kj = key[jj];
        int   ii = sid[i], ij = sid[jj];
        bool ordered = (ki > kj) || (ki == kj && ii < ij);  // stable descending
        if (descBlk ? !ordered : ordered) {
          key[i] = kj; key[jj] = ki; sid[i] = ij; sid[jj] = ii;
        }
      }
      __syncthreads();
    }
  }
  int obase = bid * NN;                 // (b,h,p) layout
  int vbase = (h * BS + b) * NN;        // (h,b,j,k) layout
  #pragma unroll
  for (int q = 0; q < 4; ++q) {
    int p = tid + q * 1024;
    int nd = sid[p];
    order[obase + p]  = nd;
    pos[obase + nd]   = p;
    val[vbase + p]    = 1.0f / (1.0f + __expf(-key[p]));   // sigmoid
  }
}

// -------- Stage 5: in-block rank (tgt = argsort(argsort(nodes))) --------
__global__ void k_tgt(const int* __restrict__ order, int* tgt) {
  __shared__ int nds[KBK];
  int bid = blockIdx.x;                 // (h*BS + b)*NB + blk
  int h = bid >> 8, b = (bid >> 5) & 7, blk = bid & 31;
  int s = threadIdx.x;
  int nd = order[(b * NH + h) * NN + blk * KBK + s];
  nds[s] = nd;
  __syncthreads();
  int r = 0;
  #pragma unroll 8
  for (int t = 0; t < KBK; ++t) r += (nds[t] < nd) ? 1 : 0;
  tgt[bid * KBK + s] = r;
}

// -------- Stage 6: build A_perm = P^T A P directly via edge scatter --------
__global__ void k_zero4(float4* p, int n4) {
  int t = blockIdx.x * blockDim.x + threadIdx.x;
  if (t < n4) { float4 z; z.x = z.y = z.z = z.w = 0.f; p[t] = z; }
}
__global__ void k_scatterA(const int* __restrict__ src, const int* __restrict__ dst,
                           const int* __restrict__ pos, const int* __restrict__ tgt,
                           const float* __restrict__ val, float* A, int E) {
  int t = blockIdx.x * blockDim.x + threadIdx.x;   // E*NH
  if (t >= E * NH) return;
  int e = t >> 2, h = t & 3;
  int sg = src[e], dg = dst[e];
  int b  = sg / NN;
  int sl = sg - b * NN;
  int dl = dg - b * NN;
  int pbase = (b * NH + h) * NN;
  int ps = pos[pbase + sl];
  int pd = pos[pbase + dl];
  if ((ps >> 7) != (pd >> 7)) return;              // keep only same-block edges
  int fb = (h * BS + b) * NB + (ps >> 7);
  int ss = ps & 127, sd = pd & 127;
  int ts = tgt[fb * KBK + ss], td = tgt[fb * KBK + sd];
  float vv = val[fb * KBK + ss] * val[fb * KBK + sd];
  atomicAdd(&A[(size_t)fb * (KBK * KBK) + ts * KBK + td], vv);
}

// -------- A_perm f32 -> packed bf16 (feeds async-to-LDS DMA in k_block) --------
__global__ void k_a2bf(const float* __restrict__ A, unsigned short* __restrict__ Ab) {
  int t = blockIdx.x * blockDim.x + threadIdx.x;   // NBLKS*KBK*KBK/4
  float4 f = ((const float4*)A)[t];
  unsigned u0 = ((unsigned)f2bf_u(f.y) << 16) | (unsigned)f2bf_u(f.x);
  unsigned u1 = ((unsigned)f2bf_u(f.w) << 16) | (unsigned)f2bf_u(f.z);
  uint2 o; o.x = u0; o.y = u1;
  ((uint2*)Ab)[t] = o;
}

// -------- W1 -> bf16 transposed (for contiguous B-fragment loads) --------
__global__ void k_w1pack(const float* __restrict__ W1, unsigned short* w1t) {
  int t = blockIdx.x * blockDim.x + threadIdx.x;   // 256*256
  int n = t >> 8, k = t & 255;
  w1t[t] = f2bf_u(W1[k * DH + n]);                 // w1t[n][k]
}

// -------- Stage 7: fused per-block dual-GEMM (bf16 WMMA) --------
#define ABF_STRIDE 136                                  // padded bf16 row stride
#define XT_STRIDE  136
#define MBF_STRIDE 264
#define XT_OFF   (128 * ABF_STRIDE * 2)                 // 34816
#define HACC_OFF (XT_OFF + 256 * XT_STRIDE * 2)         // 104448
#define SMEM_BYTES (HACC_OFF + 256 * 4)                 // 105472

__global__ void __launch_bounds__(256, 1)
k_block(const unsigned short* __restrict__ Ab, const float* __restrict__ x,
        const float* __restrict__ xa, const int* __restrict__ order,
        const int* __restrict__ tgt, const float* __restrict__ val,
        const unsigned short* __restrict__ w1t, float* __restrict__ Hacc) {
  extern __shared__ char smem[];
  // No static __shared__ in this kernel => dynamic LDS segment starts at offset 0.
  __bf16* abf = (__bf16*)(smem);               // A tile  [128][136] bf16 (byte off 0)
  __bf16* xbt = (__bf16*)(smem + XT_OFF);      // X^T tile [256][136] bf16
  __bf16* mbf = (__bf16*)(smem);               // M tile reuses LDS after GEMM1
  float*  hacc = (float*)(smem + HACC_OFF);

  const int bid = blockIdx.x;                  // (h*BS+b)*NB + blk
  const int h = bid >> 8, b = (bid >> 5) & 7, blk = bid & 31;
  const int tid = threadIdx.x;
  const int lane = tid & 31, wv = tid >> 5;    // 8 waves
  const int m = lane & 15, lh = lane >> 4;

  if (tid < 256) hacc[tid] = 0.0f;

  // Warm W1^T (bf16, 128KB) toward L2 for GEMM2 -> global_prefetch_b8
  __builtin_prefetch(w1t + (size_t)tid * 256, 0, 1);

  // ---- A tile: async DMA global(bf16) -> LDS, preserving the padded layout ----
  // 128 rows x 256B of payload = 2048 x 16B chunks; 256 threads x 8 chunks.
  {
    const unsigned short* gA = Ab + (size_t)bid * (KBK * KBK);
    #pragma unroll
    for (int q = 0; q < 8; ++q) {
      int chunk = tid + q * 256;               // 0..2047
      int r = chunk >> 4;                      // 16 chunks per 128-col row
      int c = (chunk & 15) << 3;               // starting bf16 column (8 per chunk)
      unsigned loff = (unsigned)((r * ABF_STRIDE + c) * 2);        // LDS byte offset
      unsigned long long ga = (unsigned long long)(const void*)(gA + (size_t)r * KBK + c);
      asm volatile("global_load_async_to_lds_b128 %0, %1, off"
                   :: "v"(loff), "v"(ga) : "memory");
    }
  }

  // ---- X_perm tile: gather feats, scale by val, scatter to row tgt[s], transposed ----
  {
    int s = tid >> 1, hr = tid & 1;            // 2 threads per row: x-half / x_appd-half
    int i  = tgt[bid * KBK + s];
    float vv = val[bid * KBK + s];
    int nd = order[(b * NH + h) * NN + blk * KBK + s];
    const float* rowp = (hr ? xa : x) + ((size_t)b * NN + nd) * DF;
    int nb = hr * DF;
    #pragma unroll
    for (int c = 0; c < DF; c += 4) {
      float4 f = *(const float4*)(rowp + c);
      xbt[(nb + c + 0) * XT_STRIDE + i] = f2bf(vv * f.x);
      xbt[(nb + c + 1) * XT_STRIDE + i] = f2bf(vv * f.y);
      xbt[(nb + c + 2) * XT_STRIDE + i] = f2bf(vv * f.z);
      xbt[(nb + c + 3) * XT_STRIDE + i] = f2bf(vv * f.w);
    }
  }
  asm volatile("s_wait_asynccnt 0x0" ::: "memory");  // A-tile DMA complete (this wave)
  __syncthreads();                                   // visible to all waves

  // 16-bit A-fragment K offsets per the CDNA5 16x32 layout (K pairs, split by lane half)
  int koff[8];
  #pragma unroll
  for (int v = 0; v < 8; ++v)
    koff[v] = (v < 4) ? (8 * lh + 2 * v) : (16 + 8 * lh + 2 * (v - 4));

  union Frag { v16bf v; unsigned u[8]; };
  const v8f vzero = {0.f, 0.f, 0.f, 0.f, 0.f, 0.f, 0.f, 0.f};
  v8f acc[16];
  #pragma unroll
  for (int ct = 0; ct < 16; ++ct) acc[ct] = vzero;

  const int arow = wv * 16 + m;                // this lane's A-row

  // ---- GEMM1: M(128x256) = A(128x128) @ X(128x256) ----
  for (int kk = 0; kk < KBK; kk += 32) {
    Frag fa;
    #pragma unroll
    for (int v = 0; v < 8; ++v)
      fa.u[v] = *(const unsigned*)(abf + arow * ABF_STRIDE + kk + koff[v]);
    #pragma unroll
    for (int ct = 0; ct < 16; ++ct) {
      Frag fb;
      int n = ct * 16 + m;
      #pragma unroll
      for (int v = 0; v < 8; ++v)
        fb.u[v] = *(const unsigned*)(xbt + n * XT_STRIDE + kk + koff[v]);
      acc[ct] = __builtin_amdgcn_wmma_f32_16x16x32_bf16(
          false, fa.v, false, fb.v, (short)0, acc[ct], false, false);
    }
  }
  __syncthreads();   // everyone done reading abf/xbt before LDS reuse

  // ---- store M as bf16 into reused LDS (D layout: row = r + 8*half, col = lane%16) ----
  #pragma unroll
  for (int ct = 0; ct < 16; ++ct) {
    int col = ct * 16 + m;
    #pragma unroll
    for (int r = 0; r < 8; ++r) {
      int row = wv * 16 + lh * 8 + r;
      mbf[row * MBF_STRIDE + col] = f2bf(acc[ct][r]);
    }
  }
  __syncthreads();

  // ---- GEMM2: Hh = relu(M(128x256) @ W1(256x256)) ----
  #pragma unroll
  for (int ct = 0; ct < 16; ++ct) acc[ct] = vzero;
  for (int kk = 0; kk < DH; kk += 32) {
    Frag fa;
    #pragma unroll
    for (int v = 0; v < 8; ++v)
      fa.u[v] = *(const unsigned*)(mbf + arow * MBF_STRIDE + kk + koff[v]);
    #pragma unroll
    for (int ct = 0; ct < 16; ++ct) {
      Frag fb;
      int n = ct * 16 + m;
      #pragma unroll
      for (int v = 0; v < 8; ++v)
        fb.u[v] = *(const unsigned*)(w1t + (size_t)n * DH + kk + koff[v]);
      acc[ct] = __builtin_amdgcn_wmma_f32_16x16x32_bf16(
          false, fa.v, false, fb.v, (short)0, acc[ct], false, false);
    }
  }

  // ---- relu + column-sum over the 128 rows (mean over k) ----
  #pragma unroll
  for (int ct = 0; ct < 16; ++ct) {
    float s = 0.f;
    #pragma unroll
    for (int r = 0; r < 8; ++r) s += fmaxf(acc[ct][r], 0.0f);
    s += __shfl_xor(s, 16, 32);                // merge the two row-halves
    if (lh == 0) atomicAdd(&hacc[ct * 16 + m], s);
  }
  __syncthreads();
  if (tid < 256) Hacc[(size_t)bid * DH + tid] = hacc[tid] * (1.0f / 128.0f);
}

// -------- Stage 8: head-mean, @W_out, log_softmax --------
__global__ void k_final(const float* __restrict__ Hacc, const float* __restrict__ Wout,
                        float* __restrict__ out) {
  __shared__ float hs[DH];
  __shared__ float lg[NC];
  int bid = blockIdx.x;                 // b*NB + blk
  int b = bid >> 5, blk = bid & 31;
  int tid = threadIdx.x;                // 64 threads
  #pragma unroll
  for (int q = 0; q < 4; ++q) {
    int c = tid + q * 64;
    float s = 0.f;
    #pragma unroll
    for (int hh = 0; hh < NH; ++hh)
      s += Hacc[(size_t)((hh * BS + b) * NB + blk) * DH + c];
    hs[c] = s;
  }
  __syncthreads();
  if (tid < NC) {
    float a = 0.f;
    for (int c = 0; c < DH; ++c) a += hs[c] * Wout[c * NC + tid];
    lg[tid] = 0.25f * a;                // mean over 4 heads
  }
  __syncthreads();
  if (tid < NC) {
    float mx = lg[0];
    #pragma unroll
    for (int o = 1; o < NC; ++o) mx = fmaxf(mx, lg[o]);
    float se = 0.f;
    #pragma unroll
    for (int o = 0; o < NC; ++o) se += __expf(lg[o] - mx);
    out[(size_t)bid * NC + tid] = lg[tid] - mx - __logf(se);
  }
}

extern "C" void kernel_launch(void* const* d_in, const int* in_sizes, int n_in,
                              void* d_out, int out_size, void* d_ws, size_t ws_size,
                              hipStream_t stream) {
  (void)n_in; (void)out_size; (void)ws_size;
  const float* x     = (const float*)d_in[0];
  const float* Wr    = (const float*)d_in[1];
  const float* W1    = (const float*)d_in[2];
  const float* Wout  = (const float*)d_in[3];
  const int*   eidx  = (const int*)d_in[4];
  const int E = in_sizes[4] / 2;
  const int* src = eidx;
  const int* dst = eidx + E;

  char* ws = (char*)d_ws;
  size_t off = 0;
  auto alloc = [&](size_t bytes) {
    size_t o = off; off += (bytes + 255) & ~(size_t)255; return o;
  };
  float*          deg   = (float*)(ws + alloc((size_t)NTOT * 4));
  float*          xr    = (float*)(ws + alloc((size_t)NTOT * NH * 4));
  float*          wE    = (float*)(ws + alloc((size_t)E * 4));
  float*          scs   = (float*)(ws + alloc((size_t)NTOT * NH * 4));
  int*            order = (int*)  (ws + alloc((size_t)BS * NH * NN * 4));
  int*            pos   = (int*)  (ws + alloc((size_t)BS * NH * NN * 4));
  int*            tgt   = (int*)  (ws + alloc((size_t)NBLKS * KBK * 4));
  float*          val   = (float*)(ws + alloc((size_t)NBLKS * KBK * 4));
  unsigned short* w1t   = (unsigned short*)(ws + alloc((size_t)DH * DH * 2));
  float*          HaccG = (float*)(ws + alloc((size_t)NBLKS * DH * 4));
  float*          xa    = (float*)(ws + alloc((size_t)NTOT * DF * 4));
  float*          Aperm = (float*)(ws + alloc((size_t)NBLKS * KBK * KBK * 4));
  unsigned short* Abf   = (unsigned short*)(ws + alloc((size_t)NBLKS * KBK * KBK * 2));

  k_init_deg   <<<NTOT / 256, 256, 0, stream>>>(deg);
  k_deg_scatter<<<(E + 255) / 256, 256, 0, stream>>>(dst, deg, E);
  k_rank       <<<(NTOT * NH) / 256, 256, 0, stream>>>(x, Wr, deg, xr, scs);
  k_edge_w     <<<(E + 255) / 256, 256, 0, stream>>>(src, dst, deg, wE, E);
  k_init_appd  <<<(NTOT * DF / 4) / 256, 256, 0, stream>>>(x, deg, xa);
  k_edge_feat  <<<(E * 32) / 256, 256, 0, stream>>>(src, dst, wE, x, xa, E);
  k_edge_score <<<(E + 255) / 256, 256, 0, stream>>>(src, dst, wE, xr, scs, E);
  k_sort       <<<BS * NH, 1024, 0, stream>>>(scs, order, pos, val);
  k_tgt        <<<NBLKS, KBK, 0, stream>>>(order, tgt);
  {
    int n4 = NBLKS * KBK * KBK / 4;
    k_zero4    <<<(n4 + 255) / 256, 256, 0, stream>>>((float4*)Aperm, n4);
  }
  k_scatterA   <<<(E * NH + 255) / 256, 256, 0, stream>>>(src, dst, pos, tgt, val, Aperm, E);
  k_a2bf       <<<(NBLKS * KBK * KBK / 4) / 256, 256, 0, stream>>>(Aperm, Abf);
  k_w1pack     <<<(DH * DH) / 256, 256, 0, stream>>>(W1, w1t);
  k_block      <<<NBLKS, 256, SMEM_BYTES, stream>>>(Abf, x, xa, order, tgt, val, w1t, HaccG);
  k_final      <<<BS * NB, 64, 0, stream>>>(HaccG, Wout, (float*)d_out);
}
